// DistortionAttention_54975581388902
// MI455X (gfx1250) — matrix operand order repaired and checked by reading, not compiled, not run on hardware
//
#include <hip/hip_runtime.h>

typedef __bf16 bf16_t;
typedef bf16_t v16bf __attribute__((ext_vector_type(16)));
typedef bf16_t v8bf  __attribute__((ext_vector_type(8)));
typedef float  v8f   __attribute__((ext_vector_type(8)));
typedef unsigned uv4 __attribute__((ext_vector_type(4)));
typedef unsigned uv8 __attribute__((ext_vector_type(8)));

#define NB   8
#define NC   256
#define HWD  3136     // 56*56
#define NQK  32

static __device__ __forceinline__ bf16_t f2bf(float f) {
  unsigned u = __builtin_bit_cast(unsigned, f);
  unsigned r = u + 0x7FFFu + ((u >> 16) & 1u);
  unsigned short h = (unsigned short)(r >> 16);
  return __builtin_bit_cast(bf16_t, h);
}

static __device__ __forceinline__ v8f zero8() {
  v8f z;
#pragma unroll
  for (int i = 0; i < 8; ++i) z[i] = 0.0f;
  return z;
}

static __device__ __forceinline__ v16bf zerobf() {
  v16bf z; bf16_t zb = f2bf(0.0f);
#pragma unroll
  for (int i = 0; i < 16; ++i) z[i] = zb;
  return z;
}

// A-fragment (16xK row-major source). Per ISA 7.12.2 (16-bit A 16x32):
// lane<16 (row M=lane): elems 0..7 = K 0..7, elems 8..15 = K 16..23
// lane>=16 (row M=lane-16): elems 0..7 = K 8..15, elems 8..15 = K 24..31
static __device__ __forceinline__ v16bf fragA(const bf16_t* row, int k0, int hi) {
  v8bf lo = *(const v8bf*)(row + k0 + 8 * hi);
  v8bf up = *(const v8bf*)(row + k0 + 16 + 8 * hi);
  v16bf r;
#pragma unroll
  for (int i = 0; i < 8; ++i) { r[i] = lo[i]; r[8 + i] = up[i]; }
  return r;
}

// B-fragment from Bt row-major [N][K] (B column N contiguous over K).
// lanes 0-15: elems 0..7 = K 0..7, elems 8..15 = K 16..23; lanes 16-31: K 8..15, K 24..31.
static __device__ __forceinline__ v16bf fragB(const bf16_t* rowN, int k0, int hi) {
  v8bf lo = *(const v8bf*)(rowN + k0 + 8 * hi);
  v8bf up = *(const v8bf*)(rowN + k0 + 16 + 8 * hi);
  v16bf r;
#pragma unroll
  for (int i = 0; i < 8; ++i) { r[i] = lo[i]; r[8 + i] = up[i]; }
  return r;
}

static __device__ __forceinline__ v8f wmma_bf(v16bf a, v16bf b, v8f c) {
  return __builtin_amdgcn_wmma_f32_16x16x32_bf16(false, a, false, b, (short)0, c,
                                                 false, false);
}

// TDM: async DMA of a 2-D bf16 tile [256 rows (c) x 32 cols (m)] from global
// (row stride HWD elements) into LDS, row-major compacted [c][32].
// D# group0/group1 per CDNA5 ISA 8.3/8.4; 2-D tensor -> 2 SGPR groups.
static __device__ __forceinline__ void tdm_load_vtile(const bf16_t* gsrc,
                                                      unsigned lds_off) {
  unsigned long long ga = (unsigned long long)(size_t)gsrc;
  uv4 g0;
  g0[0] = 1u;                                   // count=1, user descriptor
  g0[1] = lds_off;                              // lds_addr (bytes)
  g0[2] = (unsigned)(ga & 0xffffffffull);       // global_addr[31:0]
  g0[3] = (unsigned)((ga >> 32) & 0x1ffffffull) // global_addr[56:32]
        | (2u << 30);                           // type=2 ("image")
  uv8 g1;
  g1[0] = 1u << 16;                             // data_size=1 -> 2 bytes
  g1[1] = ((unsigned)HWD & 0xffffu) << 16;      // tensor_dim0[15:0] = 3136
  g1[2] = 256u << 16;                           // tensor_dim1 = 256
  g1[3] = 32u << 16;                            // tile_dim0 = 32
  g1[4] = 256u;                                 // tile_dim1 = 256
  g1[5] = (unsigned)HWD;                        // tensor_dim0_stride = 3136
  g1[6] = 0u;
  g1[7] = 0u;
  asm volatile("tensor_load_to_lds %0, %1" :: "s"(g0), "s"(g1) : "memory");
}

// ---------------------------------------------------------------- prep ------
__global__ void prep_kernel(const float* __restrict__ Wq, const float* __restrict__ Wk,
                            const float* __restrict__ Wv, const float* __restrict__ bq,
                            const float* __restrict__ bk, const float* __restrict__ bv,
                            const float* __restrict__ Wc1, const float* __restrict__ Wc2,
                            bf16_t* __restrict__ Wu, float* __restrict__ bu,
                            bf16_t* __restrict__ W1r, bf16_t* __restrict__ W2r,
                            float* __restrict__ pooled) {
  int stride = gridDim.x * blockDim.x;
  int tid = blockIdx.x * blockDim.x + threadIdx.x;
  for (int i = tid; i < 320 * 256; i += stride) {
    int o = i >> 8, c = i & 255;
    float w = (o < 32) ? Wq[o * 256 + c]
            : (o < 64) ? Wk[(o - 32) * 256 + c]
                       : Wv[(o - 64) * 256 + c];
    Wu[i] = f2bf(w);
  }
  for (int i = tid; i < 320; i += stride)
    bu[i] = (i < 32) ? bq[i] : (i < 64) ? bk[i - 32] : bv[i - 64];
  for (int i = tid; i < 9 * 64 * 256; i += stride) {      // W1r[tap][o][c]
    int tap = i / (64 * 256), rem = i % (64 * 256);
    int o = rem >> 8, c = rem & 255;
    W1r[i] = f2bf(Wc1[(o * 256 + c) * 9 + tap]);
  }
  for (int i = tid; i < 9 * 128 * 64; i += stride) {      // W2r[tap][o][c]
    int tap = i / (128 * 64), rem = i % (128 * 64);
    int o = rem >> 6, c = rem & 63;
    W2r[i] = f2bf(Wc2[(o * 64 + c) * 9 + tap]);
  }
  for (int i = tid; i < NB * 128; i += stride) pooled[i] = 0.0f;
}

// ----------------------------------------------------------- x transpose ----
__global__ void xt_kernel(const float* __restrict__ x, bf16_t* __restrict__ xT) {
  long total = (long)NB * NC * HWD;
  long stride = (long)gridDim.x * blockDim.x;
  for (long i = (long)blockIdx.x * blockDim.x + threadIdx.x; i < total; i += stride) {
    int c = (int)(i & 255);
    long t = i >> 8;
    int n = (int)(t % HWD);
    int b = (int)(t / HWD);
    xT[i] = f2bf(x[((long)b * NC + c) * HWD + n]);
  }
}

// ----------------------------------------------------------------- qkv ------
__global__ void qkv_kernel(const bf16_t* __restrict__ xT, const bf16_t* __restrict__ Wu,
                           const float* __restrict__ bu, bf16_t* __restrict__ q,
                           bf16_t* __restrict__ k, bf16_t* __restrict__ v) {
  int u = blockIdx.x * 8 + (threadIdx.x >> 5);   // 8*196*20 units
  int ot = u % 20;
  int t2 = u / 20;
  int nt = t2 % 196;
  int b  = t2 / 196;
  int lr = threadIdx.x & 15;
  int hi = (threadIdx.x >> 4) & 1;

  const bf16_t* Arow = Wu + (long)(ot * 16 + lr) * 256;
  const bf16_t* Brow = xT + ((long)b * HWD + nt * 16 + lr) * 256;
  v8f acc = zero8();
#pragma unroll
  for (int kc = 0; kc < 8; ++kc)
    acc = wmma_bf(fragA(Arow, kc * 32, hi), fragB(Brow, kc * 32, hi), acc);

  if (ot < 4) {  // q (ot 0-1) or k (ot 2-3), transposed layout [n][32]
    bf16_t* dst = (ot < 2) ? q : k;
    float scale = (ot < 2) ? 0.17677669529663689f : 1.0f;   // 32^-0.5 folded into q
    int obase = (ot & 1) * 16 + 8 * hi;
    v8bf pack;
#pragma unroll
    for (int r = 0; r < 8; ++r)
      pack[r] = f2bf((acc[r] + bu[ot * 16 + 8 * hi + r]) * scale);
    *(v8bf*)(dst + ((long)b * HWD + nt * 16 + lr) * 32 + obase) = pack;
  } else {       // v, natural [c][m] layout (K-contiguous for the PV GEMM)
    int c0 = ot * 16 - 64 + 8 * hi;
#pragma unroll
    for (int r = 0; r < 8; ++r) {
      float val = acc[r] + bu[ot * 16 + 8 * hi + r];
      v[((long)b * NC + c0 + r) * HWD + nt * 16 + lr] = f2bf(val);
    }
  }
}

// --------------------------------------------------------- flash attention --
// 4 waves/block, all on the same batch; each wave owns 16 query rows.
// The 32x256 bf16 v-tile is TDM-prefetched (double-buffered) into LDS once per
// block per iteration and shared by all waves.
__global__ void __launch_bounds__(128, 1)
flash_kernel(const bf16_t* __restrict__ q, const bf16_t* __restrict__ kbuf,
             const bf16_t* __restrict__ v, const float* __restrict__ x,
             float* __restrict__ out) {
  __shared__ __attribute__((aligned(64))) bf16_t vtile[2][256 * 32];  // 2 x 16KB
  __shared__ __attribute__((aligned(64))) bf16_t pS[4][16][32];       // 4 x 1KB
  int wv = threadIdx.x >> 5;
  int b  = blockIdx.x / 49;                 // all 4 waves: same batch
  int nt = (blockIdx.x % 49) * 4 + wv;      // 49*4 = 196 query tiles
  int lr = threadIdx.x & 15;
  int hi = (threadIdx.x >> 4) & 1;
  int n0 = nt * 16;

  v16bf aQ = fragA(q + ((long)b * HWD + n0 + lr) * 32, 0, hi);
  v8f acc[16];
#pragma unroll
  for (int t = 0; t < 16; ++t) acc[t] = zero8();
  float mrow[8], lrow[8];
#pragma unroll
  for (int r = 0; r < 8; ++r) { mrow[r] = -3.0e38f; lrow[r] = 0.0f; }

  const bf16_t* kbase = kbuf + (long)b * HWD * 32;
  const bf16_t* vbase = v + (long)b * NC * HWD;
  unsigned lds0 = (unsigned)(size_t)&vtile[0][0];
  unsigned lds1 = (unsigned)(size_t)&vtile[1][0];

  if (wv == 0) tdm_load_vtile(vbase, lds0);   // prefetch first tile
  int cur = 0;

  for (int m0 = 0; m0 < HWD; m0 += 32) {
    v16bf bk0 = fragB(kbase + (long)(m0 + lr) * 32, 0, hi);
    v16bf bk1 = fragB(kbase + (long)(m0 + 16 + lr) * 32, 0, hi);
    v8f s0 = wmma_bf(aQ, bk0, zero8());
    v8f s1 = wmma_bf(aQ, bk1, zero8());

    float fac[8];
#pragma unroll
    for (int r = 0; r < 8; ++r) {
      float mx = fmaxf(s0[r], s1[r]);
#pragma unroll
      for (int off = 1; off < 16; off <<= 1) mx = fmaxf(mx, __shfl_xor(mx, off, 32));
      float mnew = fmaxf(mrow[r], mx);
      fac[r] = __expf(mrow[r] - mnew);
      mrow[r] = mnew;
      float p0 = __expf(s0[r] - mnew);
      float p1 = __expf(s1[r] - mnew);
      float ps = p0 + p1;
#pragma unroll
      for (int off = 1; off < 16; off <<= 1) ps += __shfl_xor(ps, off, 32);
      lrow[r] = lrow[r] * fac[r] + ps;
      pS[wv][r + 8 * hi][lr]      = f2bf(p0);   // P tile row-major in LDS
      pS[wv][r + 8 * hi][16 + lr] = f2bf(p1);
    }
#pragma unroll
    for (int t = 0; t < 16; ++t)
#pragma unroll
      for (int r = 0; r < 8; ++r) acc[t][r] *= fac[r];

    asm volatile("s_wait_dscnt 0" ::: "memory");
    __syncthreads();                 // all waves done reading vtile[cur^1]
    if (wv == 0) {
      if (m0 + 32 < HWD) {           // prefetch next tile into the other buffer
        tdm_load_vtile(vbase + m0 + 32, cur ? lds0 : lds1);
        __builtin_amdgcn_s_wait_tensorcnt(1);   // oldest (current) tile done
      } else {
        __builtin_amdgcn_s_wait_tensorcnt(0);
      }
    }
    __syncthreads();                 // vtile[cur] visible to every wave

    v16bf aP = fragA(&pS[wv][lr][0], 0, hi);
#pragma unroll
    for (int t = 0; t < 16; ++t) {
      v16bf bV = fragB(&vtile[cur][(t * 16 + lr) * 32], 0, hi);
      acc[t] = wmma_bf(aP, bV, acc[t]);
    }
    cur ^= 1;
  }

  float inv[8];
#pragma unroll
  for (int r = 0; r < 8; ++r) inv[r] = 1.0f / lrow[r];
#pragma unroll
  for (int t = 0; t < 16; ++t) {
    long base = ((long)b * NC + t * 16 + lr) * HWD + n0 + 8 * hi;
    v8f xr = *(const v8f*)(x + base);
    v8f res;
#pragma unroll
    for (int r = 0; r < 8; ++r) res[r] = acc[t][r] * inv[r] + xr[r];
    *(v8f*)(out + base) = res;
  }
}

// --------------------------------------------------------------- conv1 ------
__global__ void conv1_kernel(const bf16_t* __restrict__ xT, const bf16_t* __restrict__ W1r,
                             const float* __restrict__ bc1, bf16_t* __restrict__ f1T) {
  int u = blockIdx.x * 8 + (threadIdx.x >> 5);   // 8*49*4 units
  int ot = u & 3;
  int t2 = u >> 2;
  int n64 = t2 % 49;
  int b   = t2 / 49;
  int lr = threadIdx.x & 15;
  int hi = (threadIdx.x >> 4) & 1;
  int n0 = n64 * 64;

  v8f acc[4];
#pragma unroll
  for (int j = 0; j < 4; ++j) acc[j] = zero8();
  v16bf zf = zerobf();

  for (int tap = 0; tap < 9; ++tap) {
    int ky = tap / 3 - 1, kx = tap % 3 - 1;
    const bf16_t* brow[4];
    bool ok[4];
#pragma unroll
    for (int j = 0; j < 4; ++j) {
      int n = n0 + j * 16 + lr;
      int h = n / 56, w = n % 56;
      int hh = h + ky, ww = w + kx;
      ok[j] = ((unsigned)hh < 56u) && ((unsigned)ww < 56u);
      int p = ok[j] ? (hh * 56 + ww) : 0;
      brow[j] = xT + ((long)b * HWD + p) * 256;
    }
    const bf16_t* arow = W1r + ((long)tap * 64 + ot * 16 + lr) * 256;
#pragma unroll
    for (int kc = 0; kc < 8; ++kc) {
      v16bf a = fragA(arow, kc * 32, hi);
#pragma unroll
      for (int j = 0; j < 4; ++j) {
        v16bf bb = fragB(brow[j], kc * 32, hi);
        bb = ok[j] ? bb : zf;
        acc[j] = wmma_bf(a, bb, acc[j]);
      }
    }
  }
#pragma unroll
  for (int j = 0; j < 4; ++j) {
    int n = n0 + j * 16 + lr;
    v8bf pack;
#pragma unroll
    for (int r = 0; r < 8; ++r)
      pack[r] = f2bf(fmaxf(acc[j][r] + bc1[ot * 16 + 8 * hi + r], 0.0f));
    *(v8bf*)(f1T + ((long)b * HWD + n) * 64 + ot * 16 + 8 * hi) = pack;
  }
}

// --------------------------------------------------------------- conv2 ------
__global__ void conv2_kernel(const bf16_t* __restrict__ f1T, const bf16_t* __restrict__ W2r,
                             const float* __restrict__ bc2, float* __restrict__ pooled) {
  int u = blockIdx.x * 8 + (threadIdx.x >> 5);   // 8*49*8 units
  int ot = u & 7;
  int t2 = u >> 3;
  int n64 = t2 % 49;
  int b   = t2 / 49;
  int lr = threadIdx.x & 15;
  int hi = (threadIdx.x >> 4) & 1;
  int n0 = n64 * 64;

  v8f acc[4];
#pragma unroll
  for (int j = 0; j < 4; ++j) acc[j] = zero8();
  v16bf zf = zerobf();

  for (int tap = 0; tap < 9; ++tap) {
    int ky = tap / 3 - 1, kx = tap % 3 - 1;
    const bf16_t* brow[4];
    bool ok[4];
#pragma unroll
    for (int j = 0; j < 4; ++j) {
      int n = n0 + j * 16 + lr;
      int h = n / 56, w = n % 56;
      int hh = h + ky, ww = w + kx;
      ok[j] = ((unsigned)hh < 56u) && ((unsigned)ww < 56u);
      int p = ok[j] ? (hh * 56 + ww) : 0;
      brow[j] = f1T + ((long)b * HWD + p) * 64;
    }
    const bf16_t* arow = W2r + ((long)tap * 128 + ot * 16 + lr) * 64;
#pragma unroll
    for (int kc = 0; kc < 2; ++kc) {
      v16bf a = fragA(arow, kc * 32, hi);
#pragma unroll
      for (int j = 0; j < 4; ++j) {
        v16bf bb = fragB(brow[j], kc * 32, hi);
        bb = ok[j] ? bb : zf;
        acc[j] = wmma_bf(a, bb, acc[j]);
      }
    }
  }
#pragma unroll
  for (int r = 0; r < 8; ++r) {
    int o = ot * 16 + 8 * hi + r;
    float s = 0.0f;
#pragma unroll
    for (int j = 0; j < 4; ++j) s += fmaxf(acc[j][r] + bc2[o], 0.0f);
#pragma unroll
    for (int off = 1; off < 16; off <<= 1) s += __shfl_xor(s, off, 32);
    if (lr == 0) atomicAdd(pooled + b * 128 + o, s * (1.0f / 3136.0f));
  }
}

// ------------------------------------------------------------------ fc ------
__global__ void fc_kernel(const float* __restrict__ pooled, const float* __restrict__ Wfc,
                          const float* __restrict__ bfc, float* __restrict__ logits) {
  int t = threadIdx.x;
  if (t < 200) {
    int b = t / 25, d = t % 25;
    float s = bfc[d];
#pragma unroll 4
    for (int kk = 0; kk < 128; ++kk) s += pooled[b * 128 + kk] * Wfc[d * 128 + kk];
    logits[b * 25 + d] = s;
  }
}

// -------------------------------------------------------------- launcher ----
extern "C" void kernel_launch(void* const* d_in, const int* in_sizes, int n_in,
                              void* d_out, int out_size, void* d_ws, size_t ws_size,
                              hipStream_t stream) {
  const float* x   = (const float*)d_in[0];
  const float* Wq  = (const float*)d_in[1];
  const float* bq  = (const float*)d_in[2];
  const float* Wk  = (const float*)d_in[3];
  const float* bk  = (const float*)d_in[4];
  const float* Wv  = (const float*)d_in[5];
  const float* bv  = (const float*)d_in[6];
  const float* Wc1 = (const float*)d_in[7];
  const float* bc1 = (const float*)d_in[8];
  const float* Wc2 = (const float*)d_in[9];
  const float* bc2 = (const float*)d_in[10];
  const float* Wfc = (const float*)d_in[11];
  const float* bfc = (const float*)d_in[12];

  char* ws = (char*)d_ws;
  size_t off = 0;
  auto carve = [&](size_t bytes) -> void* {
    void* p = ws + off;
    off = (off + bytes + 255) & ~(size_t)255;
    return p;
  };
  bf16_t* xT     = (bf16_t*)carve((size_t)NB * HWD * NC * 2);   // 12.85 MB
  bf16_t* Wu     = (bf16_t*)carve(320 * 256 * 2);
  float*  bu     = (float*) carve(320 * 4);
  bf16_t* qb     = (bf16_t*)carve((size_t)NB * HWD * NQK * 2);  // 1.6 MB
  bf16_t* kb     = (bf16_t*)carve((size_t)NB * HWD * NQK * 2);  // 1.6 MB
  bf16_t* vb     = (bf16_t*)carve((size_t)NB * NC * HWD * 2);   // 12.85 MB
  bf16_t* W1r    = (bf16_t*)carve((size_t)9 * 64 * 256 * 2);
  bf16_t* W2r    = (bf16_t*)carve((size_t)9 * 128 * 64 * 2);
  bf16_t* f1T    = (bf16_t*)carve((size_t)NB * HWD * 64 * 2);   // 3.2 MB
  float*  pooled = (float*) carve((size_t)NB * 128 * 4);
  (void)ws_size; (void)in_sizes; (void)n_in; (void)out_size;

  float* attn_out = (float*)d_out;
  float* logits   = (float*)d_out + (size_t)NB * NC * HWD;   // 6,422,528

  prep_kernel<<<128, 256, 0, stream>>>(Wq, Wk, Wv, bq, bk, bv, Wc1, Wc2,
                                       Wu, bu, W1r, W2r, pooled);
  xt_kernel<<<8192, 256, 0, stream>>>(x, xT);
  qkv_kernel<<<3920, 256, 0, stream>>>(xT, Wu, bu, qb, kb, vb);   // 8*196*20 waves
  conv1_kernel<<<196, 256, 0, stream>>>(xT, W1r, bc1, f1T);       // 8*49*4 waves
  flash_kernel<<<392, 128, 0, stream>>>(qb, kb, vb, x, attn_out); // 8*49 blocks*4 waves
  conv2_kernel<<<392, 256, 0, stream>>>(f1T, W2r, bc2, pooled);   // 8*49*8 waves
  fc_kernel<<<1, 256, 0, stream>>>(pooled, Wfc, bfc, logits);
}
